// LSHConv_22308060135934
// MI455X (gfx1250) — matrix operand order repaired and checked by reading, not compiled, not run on hardware
//
#include <hip/hip_runtime.h>
#include <hip/hip_bf16.h>
#include <math.h>

// Problem constants (from reference)
#define BB 4
#define LL 8192
#define DD 1024
#define HH 8
#define DH 128
#define EPSV 0.0001f
#define CHUNK 256   // sorted positions per block in conv kernel (16 tiles of 16)

typedef __attribute__((ext_vector_type(16))) __bf16 v16bf;
typedef __attribute__((ext_vector_type(8)))  float  v8f;

union ABFrag {
    v16bf v;
    uint4 q[2];          // two 16-byte LDS loads fill a fragment
};

__device__ __forceinline__ unsigned short f2bf(float f) {
    unsigned int x = __float_as_uint(f);
    // round-to-nearest-even bf16 truncation
    unsigned int r = (x + 0x7FFFu + ((x >> 16) & 1u)) >> 16;
    return (unsigned short)r;
}

// ---------------------------------------------------------------------------
// Kernel 1: per-(b,l,h) 128->2 projection + arctan angle.
// grid = B*L blocks of 256 threads; wave w handles head w (wave32).
// ---------------------------------------------------------------------------
__global__ void hash_angles_kernel(const float* __restrict__ x,
                                   const float* __restrict__ hash_w,
                                   const float* __restrict__ hash_b,
                                   float* __restrict__ angles) {
    const int bl   = blockIdx.x;            // 0 .. B*L-1
    const int b    = bl / LL;
    const int l    = bl % LL;
    const int h    = threadIdx.x >> 5;      // wave id = head
    const int lane = threadIdx.x & 31;

    const float* xrow = x + ((size_t)bl) * DD + h * DH;
    const float* wrow = hash_w + (size_t)h * DH * 2;

    float s0 = 0.f, s1 = 0.f;
#pragma unroll
    for (int k = 0; k < DH / 32; ++k) {
        int d = lane + k * 32;
        float xv = xrow[d];
        s0 += xv * wrow[d * 2 + 0];
        s1 += xv * wrow[d * 2 + 1];
    }
#pragma unroll
    for (int off = 16; off > 0; off >>= 1) {
        s0 += __shfl_xor(s0, off, 32);
        s1 += __shfl_xor(s1, off, 32);
    }
    if (lane == 0) {
        float hx = s0 + hash_b[h * 2 + 0];
        float hy = s1 + hash_b[h * 2 + 1];
        angles[((size_t)(b * HH + h)) * LL + l] = atanf(hx / (hy + EPSV));
    }
}

// ---------------------------------------------------------------------------
// Kernel 2: bitonic argsort of 8192 angles per (b,h) in LDS.
// Tie-break on original index -> matches stable jnp.argsort.
// grid = B*H blocks of 1024 threads.
// ---------------------------------------------------------------------------
__global__ void bitonic_argsort_kernel(const float* __restrict__ angles,
                                       int* __restrict__ idx_out) {
    __shared__ float key[LL];
    __shared__ int   val[LL];

    const size_t base = (size_t)blockIdx.x * LL;   // blockIdx.x = b*H + h
    for (int i = threadIdx.x; i < LL; i += 1024) {
        key[i] = angles[base + i];
        val[i] = i;
    }
    __syncthreads();

    for (int k = 2; k <= LL; k <<= 1) {
        for (int j = k >> 1; j > 0; j >>= 1) {
            for (int i = threadIdx.x; i < LL; i += 1024) {
                int ij = i ^ j;
                if (ij > i) {
                    float ka = key[i], kb = key[ij];
                    int   va = val[i], vb = val[ij];
                    bool up = ((i & k) == 0);
                    // lexicographic (key, idx) greater-than
                    bool aGTb = (ka > kb) || (ka == kb && va > vb);
                    bool doSwap = up ? aGTb : !aGTb;
                    if (doSwap) {
                        key[i] = kb; key[ij] = ka;
                        val[i] = vb; val[ij] = va;
                    }
                }
            }
            __syncthreads();
        }
    }
    for (int i = threadIdx.x; i < LL; i += 1024)
        idx_out[base + i] = val[i];
}

// ---------------------------------------------------------------------------
// Kernel 3: gathered grouped conv as bf16 WMMA + fused bias + inverse-perm
// scatter. grid = (L/CHUNK, H, B), 256 threads (8 waves).
// Wave w computes output channels [16w, 16w+16) of its head.
// Y(16 x 128) = X(16 x 384) @ W(384 x 128), K split into 12 steps of 32.
// Fragments are built with 16-byte ds loads (layouts are per-lane contiguous).
// ---------------------------------------------------------------------------
__global__ void lsh_conv_wmma_kernel(const float* __restrict__ x,
                                     const float* __restrict__ conv_w,
                                     const float* __restrict__ conv_b,
                                     const int* __restrict__ idx,
                                     float* __restrict__ out) {
    // Per-head weights, bf16, layout [tap][o][i]  (3*128*128 = 96 KB)
    __shared__ unsigned short WLT[3 * DH * DH];
    // 18 gathered token rows (16 outputs + 1 halo each side), bf16 [row][i]
    __shared__ unsigned short XS[18 * DH];

    const int b = blockIdx.z;
    const int h = blockIdx.y;
    const int chunk0 = blockIdx.x * CHUNK;

    const int lane = threadIdx.x & 31;
    const int w    = threadIdx.x >> 5;
    const int m16  = lane & 15;
    const int half = lane >> 4;
    const int obase = w * 16;

    const int* idxp = idx + ((size_t)(b * HH + h)) * LL;

    // Stage per-head weights: WLT[t][o][i] = conv_w[(h*128+o)*128 + i][t]
    for (int n = threadIdx.x; n < 3 * DH * DH; n += 256) {
        int i = n & 127;
        int o = (n >> 7) & 127;
        int t = n >> 14;
        float wv = conv_w[(((size_t)(h * DH + o)) * DH + i) * 3 + t];
        WLT[n] = f2bf(wv);
    }
    __syncthreads();

    for (int tile = 0; tile < CHUNK / 16; ++tile) {
        const int p0 = chunk0 + tile * 16;

        __syncthreads();   // protect XS from previous tile's readers
        // Gather 18 sorted rows into LDS (row r = sorted pos (p0-1+r) mod L)
        for (int n = threadIdx.x; n < 18 * DH; n += 256) {
            int r = n >> 7;
            int i = n & 127;
            int sp = p0 - 1 + r;
            sp = (sp + LL) % LL;
            int orig = idxp[sp];
            float xv = x[((size_t)(b * LL + orig)) * DD + h * DH + i];
            XS[n] = f2bf(xv);
        }
        __syncthreads();

        v8f acc = {};
#pragma unroll
        for (int kk = 0; kk < 12; ++kk) {
            const int t  = kk >> 2;          // tap 0..2
            const int cb = (kk & 3) * 32;    // channel base within tap

            ABFrag a, bm;
            // --- A fragment: 16x32 bf16. Lane: M = lane&15.
            // V0-7 hold K = half*16 + [0..15]  (32 contiguous bytes)
            const unsigned short* xrow = &XS[(m16 + t) * DH + cb];
            a.q[0] = *(const uint4*)(xrow + half * 16);
            a.q[1] = *(const uint4*)(xrow + half * 16 + 8);

            // --- B fragment: 32x16 bf16. Lane: N = lane&15.
            // V0-7 hold K = half*16 + [0..15]; WLT[t][o][i] keeps K-pairs
            // contiguous along i -> two 16-byte loads.
            const unsigned short* wrow =
                &WLT[((size_t)t * DH + obase + m16) * DH + cb + half * 16];
            bm.q[0] = *(const uint4*)(wrow);
            bm.q[1] = *(const uint4*)(wrow + 8);

            acc = __builtin_amdgcn_wmma_f32_16x16x32_bf16(
                /*neg_a=*/false, a.v, /*neg_b=*/false, bm.v,
                /*c_mod=*/(short)0, acc, /*reuse_a=*/false, /*reuse_b=*/false);
        }

        // Epilogue: C/D layout -> lane: N = lane&15, VGPR r: M = r + 8*half.
        // Fused inverse-permutation scatter: out[b, idx[p], h*128+o] = y + bias
        const int o = obase + m16;
        const float bias = conv_b[h * DH + o];
#pragma unroll
        for (int r = 0; r < 8; ++r) {
            int M = r + 8 * half;
            int orig = idxp[p0 + M];
            out[((size_t)(b * LL + orig)) * DD + h * DH + o] = acc[r] + bias;
        }
    }
}

// ---------------------------------------------------------------------------
extern "C" void kernel_launch(void* const* d_in, const int* in_sizes, int n_in,
                              void* d_out, int out_size, void* d_ws, size_t ws_size,
                              hipStream_t stream) {
    const float* x      = (const float*)d_in[0];
    const float* hash_w = (const float*)d_in[1];
    const float* hash_b = (const float*)d_in[2];
    const float* conv_w = (const float*)d_in[3];
    const float* conv_b = (const float*)d_in[4];
    float* out = (float*)d_out;

    // Workspace layout: angles (B*H*L f32, 1 MB) | idx (B*H*L i32, 1 MB)
    float* angles = (float*)d_ws;
    int*   idx    = (int*)((char*)d_ws + (size_t)BB * HH * LL * sizeof(float));

    hash_angles_kernel<<<BB * LL, 256, 0, stream>>>(x, hash_w, hash_b, angles);
    bitonic_argsort_kernel<<<BB * HH, 1024, 0, stream>>>(angles, idx);
    lsh_conv_wmma_kernel<<<dim3(LL / CHUNK, HH, BB), 256, 0, stream>>>(
        x, conv_w, conv_b, idx, out);
}
